// VocabGraphAttention_12876311953624
// MI455X (gfx1250) — matrix operand order repaired and checked by reading, not compiled, not run on hardware
//
#include <hip/hip_runtime.h>
#include <hip/hip_bf16.h>

// ---------------- problem constants ----------------
static constexpr int NV       = 30000;
static constexpr int NE       = 480000;
static constexpr int HID      = 256;
static constexpr int HEADS    = 4;
static constexpr int HEAD_DIM = 64;
static constexpr int MROWS    = 1024;      // 2 * 512 batch rows
static constexpr float NEG_SLOPE = 0.2f;
static constexpr float EPS       = 1e-9f;

typedef __attribute__((ext_vector_type(16))) __bf16 v16bf;
typedef __attribute__((ext_vector_type(8)))  float  v8f;

// ---------------- utility ----------------
__global__ void vga_zero(float* __restrict__ p, int n) {
  int i = blockIdx.x * blockDim.x + threadIdx.x;
  if (i < n) p[i] = 0.0f;
}

// H[src] += val * W[dst]   (one wave per edge, 8 cols per lane)
__global__ __launch_bounds__(256) void vga_scatter(
    const float* __restrict__ W, const float* __restrict__ val,
    const int* __restrict__ src, const int* __restrict__ dst,
    float* __restrict__ H) {
  int e = (blockIdx.x * blockDim.x + threadIdx.x) >> 5;
  if (e >= NE) return;
  int lane = threadIdx.x & 31;
  float ve = val[e];
  const float* wrow = W + (long)dst[e] * HID;
  float*       hrow = H + (long)src[e] * HID;
#pragma unroll
  for (int i = 0; i < HID / 32; ++i) {
    int c = lane + 32 * i;
    atomicAdd(&hrow[c], ve * wrow[c]);
  }
}

// per (edge, head): e = leakyrelu(concat(H[src],H[dst]) . a[h]); eexp stored; denom[dst] += exp(e)
__global__ __launch_bounds__(256) void vga_scores(
    const float* __restrict__ H, const float* __restrict__ a,
    const int* __restrict__ src, const int* __restrict__ dst,
    float* __restrict__ eexp, float* __restrict__ denom) {
  int idx = blockIdx.x * blockDim.x + threadIdx.x;
  if (idx >= NE * HEADS) return;
  int e = idx >> 2;
  int h = idx & 3;
  const float* hs = H + (long)src[e] * HID + h * HEAD_DIM;
  const float* hd = H + (long)dst[e] * HID + h * HEAD_DIM;
  const float* ah = a + h * (2 * HEAD_DIM);
  float s = 0.0f;
#pragma unroll 8
  for (int d = 0; d < HEAD_DIM; ++d)
    s += hs[d] * ah[d] + hd[d] * ah[HEAD_DIM + d];
  s = (s > 0.0f) ? s : NEG_SLOPE * s;
  float ex = expf(s);
  eexp[idx] = ex;
  atomicAdd(&denom[(long)dst[e] * HEADS + h], ex);
}

// Hsum[dst] += (eexp / (denom[dst]+eps)) * H[src]   (one wave per edge)
__global__ __launch_bounds__(256) void vga_agg(
    const float* __restrict__ H, const float* __restrict__ eexp,
    const float* __restrict__ denom, const int* __restrict__ src,
    const int* __restrict__ dst, float* __restrict__ Hsum) {
  int e = (blockIdx.x * blockDim.x + threadIdx.x) >> 5;
  if (e >= NE) return;
  int lane = threadIdx.x & 31;
  int de = dst[e];
  const float* hrow = H + (long)src[e] * HID;
  float*       orow = Hsum + (long)de * HID;
  float al[HEADS];
#pragma unroll
  for (int h = 0; h < HEADS; ++h)
    al[h] = eexp[(long)e * HEADS + h] / (denom[(long)de * HEADS + h] + EPS);
#pragma unroll
  for (int i = 0; i < HID / 32; ++i) {
    int c = lane + 32 * i;
    atomicAdd(&orow[c], al[c >> 6] * hrow[c]);
  }
}

// HT[h][v] = bf16(Hsum[v][h])  -- transposed bf16 so GEMM B loads are K-contiguous
__global__ __launch_bounds__(256) void vga_transpose(
    const float* __restrict__ Hsum, __bf16* __restrict__ HT) {
  int idx = blockIdx.x * blockDim.x + threadIdx.x;
  if (idx >= NV * HID) return;
  int h = idx / NV;
  int v = idx - h * NV;
  HT[idx] = (__bf16)Hsum[(long)v * HID + h];
}

// ---------------- WMMA fragment K-offset patterns (ISA 7.12.2, wave32) ----------------
// 16-bit A 16x32: VGPR v<4 -> K=2v(,+1), v>=4 -> K=2v+8; lanes 16-31 add +8
__device__ __forceinline__ int kpatA(int v, int half) {
  return ((v < 4) ? (2 * v) : (2 * v + 8)) + 8 * half;
}
// 16-bit B 32x16: same K order as A: VGPR v<4 -> K=2v(,+1), v>=4 -> K=2v+8; lanes 16-31 add +8
__device__ __forceinline__ int kpatB(int v, int half) {
  return ((v < 4) ? (2 * v) : (2 * v + 8)) + 8 * half;
}

// C[1024,256] = X[1024,30000](f32->bf16) * Hsum  with B supplied as HT[256][30000] bf16.
// grid (32, 2), block 128 (4 waves). Wave computes a 32(M) x 32(N) tile = 2x2 WMMA tiles.
__global__ __launch_bounds__(128) void vga_gemm_xh(
    const float* __restrict__ X, const __bf16* __restrict__ HT,
    float* __restrict__ C) {
  const int lane = threadIdx.x & 31;
  const int wave = threadIdx.x >> 5;   // 0..3
  const int row  = lane & 15;
  const int half = lane >> 4;
  const int mbase = blockIdx.x * 32;
  const int nbase = blockIdx.y * 128 + wave * 32;
  const float*  A0 = X  + (long)(mbase + row) * NV;
  const float*  A1 = A0 + (long)16 * NV;
  const __bf16* B0 = HT + (long)(nbase + row) * NV;
  const __bf16* B1 = B0 + (long)16 * NV;

  v8f acc00 = {}, acc01 = {}, acc10 = {}, acc11 = {};

  int kb = 0;
  for (; kb + 32 <= NV; kb += 32) {
    v16bf a0, a1, b0, b1;
#pragma unroll
    for (int v = 0; v < 8; ++v) {
      int ka = kb + kpatA(v, half);
      a0[2 * v]     = (__bf16)A0[ka];
      a0[2 * v + 1] = (__bf16)A0[ka + 1];
      a1[2 * v]     = (__bf16)A1[ka];
      a1[2 * v + 1] = (__bf16)A1[ka + 1];
      int kc = kb + kpatB(v, half);
      b0[2 * v]     = B0[kc];
      b0[2 * v + 1] = B0[kc + 1];
      b1[2 * v]     = B1[kc];
      b1[2 * v + 1] = B1[kc + 1];
    }
    acc00 = __builtin_amdgcn_wmma_f32_16x16x32_bf16(false, a0, false, b0, (short)0, acc00, false, false);
    acc01 = __builtin_amdgcn_wmma_f32_16x16x32_bf16(false, a0, false, b1, (short)0, acc01, false, false);
    acc10 = __builtin_amdgcn_wmma_f32_16x16x32_bf16(false, a1, false, b0, (short)0, acc10, false, false);
    acc11 = __builtin_amdgcn_wmma_f32_16x16x32_bf16(false, a1, false, b1, (short)0, acc11, false, false);
  }
  // tail: NV % 32 == 16 -> one guarded step (pairs never straddle the boundary)
  {
    v16bf a0 = {}, a1 = {}, b0 = {}, b1 = {};
#pragma unroll
    for (int v = 0; v < 8; ++v) {
      int ka = kb + kpatA(v, half);
      if (ka < NV) {
        a0[2 * v]     = (__bf16)A0[ka];
        a0[2 * v + 1] = (__bf16)A0[ka + 1];
        a1[2 * v]     = (__bf16)A1[ka];
        a1[2 * v + 1] = (__bf16)A1[ka + 1];
      }
      int kc = kb + kpatB(v, half);
      if (kc < NV) {
        b0[2 * v]     = B0[kc];
        b0[2 * v + 1] = B0[kc + 1];
        b1[2 * v]     = B1[kc];
        b1[2 * v + 1] = B1[kc + 1];
      }
    }
    acc00 = __builtin_amdgcn_wmma_f32_16x16x32_bf16(false, a0, false, b0, (short)0, acc00, false, false);
    acc01 = __builtin_amdgcn_wmma_f32_16x16x32_bf16(false, a0, false, b1, (short)0, acc01, false, false);
    acc10 = __builtin_amdgcn_wmma_f32_16x16x32_bf16(false, a1, false, b0, (short)0, acc10, false, false);
    acc11 = __builtin_amdgcn_wmma_f32_16x16x32_bf16(false, a1, false, b1, (short)0, acc11, false, false);
  }
  // C/D layout: VGPR r -> M = r + 8*half, N = lane%16
#pragma unroll
  for (int r = 0; r < 8; ++r) {
    int M0 = mbase + r + 8 * half;
    int N0 = nbase + row;
    C[(long)M0 * HID + N0]             = acc00[r];
    C[(long)M0 * HID + N0 + 16]        = acc01[r];
    C[(long)(M0 + 16) * HID + N0]      = acc10[r];
    C[(long)(M0 + 16) * HID + N0 + 16] = acc11[r];
  }
}

// out[1024,256] = fused[1024,256] @ fc_w^T + fc_b.  B[k][n] = fc_w[n*256+k] (rows K-contiguous).
__global__ __launch_bounds__(128) void vga_gemm_fc(
    const float* __restrict__ A, const float* __restrict__ Wf,
    const float* __restrict__ bias, float* __restrict__ out) {
  const int lane = threadIdx.x & 31;
  const int wave = threadIdx.x >> 5;
  const int row  = lane & 15;
  const int half = lane >> 4;
  const int mbase = blockIdx.x * 32;
  const int nbase = blockIdx.y * 128 + wave * 32;
  const float* A0 = A  + (long)(mbase + row) * HID;
  const float* A1 = A0 + 16 * HID;
  const float* B0 = Wf + (long)(nbase + row) * HID;
  const float* B1 = B0 + 16 * HID;

  v8f acc00 = {}, acc01 = {}, acc10 = {}, acc11 = {};
#pragma unroll 2
  for (int kb = 0; kb < HID; kb += 32) {
    v16bf a0, a1, b0, b1;
#pragma unroll
    for (int v = 0; v < 8; ++v) {
      int ka = kb + kpatA(v, half);
      a0[2 * v]     = (__bf16)A0[ka];
      a0[2 * v + 1] = (__bf16)A0[ka + 1];
      a1[2 * v]     = (__bf16)A1[ka];
      a1[2 * v + 1] = (__bf16)A1[ka + 1];
      int kc = kb + kpatB(v, half);
      b0[2 * v]     = (__bf16)B0[kc];
      b0[2 * v + 1] = (__bf16)B0[kc + 1];
      b1[2 * v]     = (__bf16)B1[kc];
      b1[2 * v + 1] = (__bf16)B1[kc + 1];
    }
    acc00 = __builtin_amdgcn_wmma_f32_16x16x32_bf16(false, a0, false, b0, (short)0, acc00, false, false);
    acc01 = __builtin_amdgcn_wmma_f32_16x16x32_bf16(false, a0, false, b1, (short)0, acc01, false, false);
    acc10 = __builtin_amdgcn_wmma_f32_16x16x32_bf16(false, a1, false, b0, (short)0, acc10, false, false);
    acc11 = __builtin_amdgcn_wmma_f32_16x16x32_bf16(false, a1, false, b1, (short)0, acc11, false, false);
  }
  float bN0  = bias[nbase + row];
  float bN16 = bias[nbase + row + 16];
#pragma unroll
  for (int r = 0; r < 8; ++r) {
    int M0 = mbase + r + 8 * half;
    int N0 = nbase + row;
    out[(long)M0 * HID + N0]             = acc00[r] + bN0;
    out[(long)M0 * HID + N0 + 16]        = acc01[r] + bN16;
    out[(long)(M0 + 16) * HID + N0]      = acc10[r] + bN0;
    out[(long)(M0 + 16) * HID + N0 + 16] = acc11[r] + bN16;
  }
}

// ---------------- host launcher ----------------
extern "C" void kernel_launch(void* const* d_in, const int* in_sizes, int n_in,
                              void* d_out, int out_size, void* d_ws, size_t ws_size,
                              hipStream_t stream) {
  (void)in_sizes; (void)n_in; (void)out_size; (void)ws_size;
  const float* X    = (const float*)d_in[0];
  const float* W0   = (const float*)d_in[1];
  const float* a0   = (const float*)d_in[2];
  const float* W1   = (const float*)d_in[3];
  const float* a1   = (const float*)d_in[4];
  const float* fc_w = (const float*)d_in[5];
  const float* fc_b = (const float*)d_in[6];
  const float* av0  = (const float*)d_in[7];
  const float* av1  = (const float*)d_in[8];
  const int*   s0   = (const int*)d_in[9];
  const int*   dd0  = (const int*)d_in[10];
  const int*   s1   = (const int*)d_in[11];
  const int*   dd1  = (const int*)d_in[12];
  float* out = (float*)d_out;

  // workspace layout (~70 MB): [Hsum f32][H f32][denom f32][eexp f32]
  // HT (bf16) and fused (f32) alias the dead H region after the graph passes.
  float* Hsum  = (float*)d_ws;
  float* H     = Hsum + (size_t)NV * HID;
  float* denom = H + (size_t)NV * HID;
  float* eexp  = denom + (size_t)NV * HEADS;
  __bf16* HT   = (__bf16*)H;                         // NV*HID bf16 = half of H region
  float* fused = H + (size_t)NV * HID / 2;           // next 1 MB of H region

  const int nzero1 = NV * HID;                       // Hsum
  const int nzero2 = NV * HID + NV * HEADS;          // H + denom (contiguous)
  const int scatterBlocks = (NE * 32) / 256;         // 60000
  const int scoreBlocks   = (NE * HEADS + 255) / 256;
  const int tposeBlocks   = (NV * HID + 255) / 256;

  vga_zero<<<(nzero1 + 255) / 256, 256, 0, stream>>>(Hsum, nzero1);

  // graph 0
  vga_zero<<<(nzero2 + 255) / 256, 256, 0, stream>>>(H, nzero2);
  vga_scatter<<<scatterBlocks, 256, 0, stream>>>(W0, av0, s0, dd0, H);
  vga_scores<<<scoreBlocks, 256, 0, stream>>>(H, a0, s0, dd0, eexp, denom);
  vga_agg<<<scatterBlocks, 256, 0, stream>>>(H, eexp, denom, s0, dd0, Hsum);

  // graph 1
  vga_zero<<<(nzero2 + 255) / 256, 256, 0, stream>>>(H, nzero2);
  vga_scatter<<<scatterBlocks, 256, 0, stream>>>(W1, av1, s1, dd1, H);
  vga_scores<<<scoreBlocks, 256, 0, stream>>>(H, a1, s1, dd1, eexp, denom);
  vga_agg<<<scatterBlocks, 256, 0, stream>>>(H, eexp, denom, s1, dd1, Hsum);

  // transpose+convert, then WMMA GEMMs
  vga_transpose<<<tposeBlocks, 256, 0, stream>>>(Hsum, HT);
  vga_gemm_xh<<<dim3(MROWS / 32, HID / 128), 128, 0, stream>>>(X, HT, fused);
  vga_gemm_fc<<<dim3(MROWS / 32, HID / 128), 128, 0, stream>>>(fused, fc_w, fc_b, out);
}
